// PivotalAttentionBlock_36541581754855
// MI455X (gfx1250) — matrix operand order repaired and checked by reading, not compiled, not run on hardware
//
#include <hip/hip_runtime.h>

// CDNA5 / gfx1250, wave32. Matrix math via v_wmma_f32_16x16x32_bf16;
// global->LDS staging via GLOBAL_LOAD_ASYNC_TO_LDS_B128 (ASYNCcnt) when available.

typedef __bf16 bf16;
typedef __attribute__((ext_vector_type(16))) __bf16 v16bf;
typedef __attribute__((ext_vector_type(8)))  float  v8f;
typedef __attribute__((ext_vector_type(4)))  int    v4i;

#define WMMA(a, b, c) \
  __builtin_amdgcn_wmma_f32_16x16x32_bf16(false, (a), false, (b), (short)0, (c), false, false)

#define B_  2
#define L_  128
#define D_  256
#define H_  8
#define HD_ 32
#define M_  (B_ * L_ * L_)   // 32768 tokens

// ---------------- async global->LDS copy (16B per lane) ----------------
#if defined(__HIP_DEVICE_COMPILE__) && \
    __has_builtin(__builtin_amdgcn_global_load_async_to_lds_b128)
#define HAVE_ASYNC 1
#else
#define HAVE_ASYNC 0
#endif

#define AS1 __attribute__((address_space(1)))
#define AS3 __attribute__((address_space(3)))

static __device__ __forceinline__ void lds_cp16(const bf16* __restrict__ g,
                                                bf16* __restrict__ l) {
#if HAVE_ASYNC
  __builtin_amdgcn_global_load_async_to_lds_b128((AS1 v4i*)g, (AS3 v4i*)l, 0, 0);
#else
  *(uint4*)l = *(const uint4*)g;
#endif
}
static __device__ __forceinline__ void async_wait0() {
#if HAVE_ASYNC
#if __has_builtin(__builtin_amdgcn_s_wait_asynccnt)
  __builtin_amdgcn_s_wait_asynccnt(0);
#else
  asm volatile("s_wait_asynccnt 0x0" ::: "memory");
#endif
#endif
}

// ---- WMMA fragment loaders from row-major LDS tiles (row = m or n, col = k) ----
// A 16x32 bf16 layout: lane l holds row m=l&15; VGPR v<4: k = 8*(l>>4)+2v(+1);
//                      VGPR v>=4: k = 16 + 8*(l>>4) + 2(v-4)(+1).
static __device__ __forceinline__ v16bf frag_a(const bf16* __restrict__ t, int ldk,
                                               int row0, int koff, int lane) {
  const bf16* p = t + (size_t)(row0 + (lane & 15)) * ldk + koff;
  const int hi8 = (lane >> 4) << 3;
  v16bf f;
#pragma unroll
  for (int v = 0; v < 4; ++v) {
    f[2 * v]     = p[hi8 + 2 * v];
    f[2 * v + 1] = p[hi8 + 2 * v + 1];
    f[8 + 2 * v]     = p[16 + hi8 + 2 * v];
    f[8 + 2 * v + 1] = p[16 + hi8 + 2 * v + 1];
  }
  return f;
}
// B 32x16 bf16 layout: lane l holds col n=l&15; same k order as A:
// elems 0-7: k = 8*(l>>4)+s; elems 8-15: k = 16 + 8*(l>>4) + (s-8). Tile stored as B^T.
static __device__ __forceinline__ v16bf frag_b(const bf16* __restrict__ t, int ldk,
                                               int col0, int koff, int lane) {
  const bf16* p = t + (size_t)(col0 + (lane & 15)) * ldk + koff + ((lane >> 4) << 3);
  v16bf f;
#pragma unroll
  for (int v = 0; v < 16; ++v) f[v] = p[((v & 8) << 1) + (v & 7)];
  return f;
}

// =====================================================================
// Generic tiled GEMM:  C[m,n] = epilogue( sum_k A[m,k]*Bt[n,k] )
// BM=128, BN=64, KC=128, double-buffered async LDS staging.
// 256 threads = 8 waves; wave w does rows 16w..16w+15, 4 col-subtiles of 16.
// =====================================================================
template <typename OutT, bool RELU, bool RESID, bool SCALE>
__global__ __launch_bounds__(256) void gemm_bf16_kernel(
    const bf16* __restrict__ A, const bf16* __restrict__ Bt, OutT* __restrict__ C,
    const float* __restrict__ resid, const float* __restrict__ scale_ptr,
    int M, int N, int K) {
  __shared__ __align__(16) bf16 sA[2][128 * 128];
  __shared__ __align__(16) bf16 sB[2][64 * 128];
  const int tid = threadIdx.x, lane = tid & 31, wv = tid >> 5;
  const int m0 = blockIdx.y * 128, n0 = blockIdx.x * 64;
  v8f acc[4] = {};

  auto stage = [&](int buf, int kc) {
#pragma unroll
    for (int ii = 0; ii < 8; ++ii) {   // A panel: 128x128 bf16 = 2048 x 16B
      int c = tid + 256 * ii, r = c >> 4, cc = c & 15;
      lds_cp16(&A[(size_t)(m0 + r) * K + kc + cc * 8], &sA[buf][r * 128 + cc * 8]);
    }
#pragma unroll
    for (int ii = 0; ii < 4; ++ii) {   // B panel: 64x128 bf16 = 1024 x 16B
      int c = tid + 256 * ii, r = c >> 4, cc = c & 15;
      lds_cp16(&Bt[(size_t)(n0 + r) * K + kc + cc * 8], &sB[buf][r * 128 + cc * 8]);
    }
  };

  stage(0, 0);
  async_wait0();
  __syncthreads();
  int cur = 0;
  for (int kc = 0; kc < K; kc += 128) {
    const bool more = (kc + 128) < K;
    if (more) stage(cur ^ 1, kc + 128);   // overlap next fetch with WMMA
#pragma unroll
    for (int kk = 0; kk < 128; kk += 32) {
      v16bf fa = frag_a(sA[cur], 128, wv * 16, kk, lane);
#pragma unroll
      for (int c = 0; c < 4; ++c) {
        v16bf fb = frag_b(sB[cur], 128, c * 16, kk, lane);
        acc[c] = WMMA(fa, fb, acc[c]);
      }
    }
    if (more) {
      async_wait0();
      __syncthreads();
      cur ^= 1;
    }
  }
  const float s = SCALE ? *scale_ptr : 1.0f;
  const int nn = lane & 15, hi = lane >> 4;
#pragma unroll
  for (int c = 0; c < 4; ++c) {
    int gn = n0 + c * 16 + nn;
#pragma unroll
    for (int r = 0; r < 8; ++r) {
      int gm = m0 + wv * 16 + hi * 8 + r;
      float v = acc[c][r];
      if (SCALE) v *= s;
      if (RELU)  v = v > 0.0f ? v : 0.0f;
      size_t o = (size_t)gm * N + gn;
      if (RESID) v += resid[o];
      C[o] = (OutT)v;
    }
  }
}

// ---- elementwise helpers ----
__global__ __launch_bounds__(256) void f32_to_bf16_kernel(const float* __restrict__ in,
                                                          bf16* __restrict__ out, int n) {
  int i = blockIdx.x * 256 + threadIdx.x;
  if (i < n) out[i] = (bf16)in[i];
}

// xbt[b,i,k,:] = bf16(x[b,k,i,:])   (rows for the symmetric-mix GEMM)
__global__ __launch_bounds__(256) void xpose_x_kernel(const float* __restrict__ x,
                                                      bf16* __restrict__ out) {
  size_t idx = (size_t)blockIdx.x * 256 + threadIdx.x;
  int d = (int)(idx & 255);
  size_t r = idx >> 8;
  int k = (int)(r & 127), i = (int)((r >> 7) & 127), b = (int)(r >> 14);
  out[idx] = (bf16)x[(((size_t)b * 128 + k) * 128 + i) * 256 + d];
}

// row layernorm: one wave per 256-wide row, bf16 out
__global__ __launch_bounds__(256) void layernorm_kernel(const float* __restrict__ X,
                                                        bf16* __restrict__ Y,
                                                        const float* __restrict__ g,
                                                        const float* __restrict__ b) {
  int row = blockIdx.x * 8 + (threadIdx.x >> 5);
  int lane = threadIdx.x & 31;
  const float* x = X + (size_t)row * 256;
  float v[8], s = 0.f;
#pragma unroll
  for (int q = 0; q < 8; ++q) { v[q] = x[lane + 32 * q]; s += v[q]; }
#pragma unroll
  for (int o = 16; o; o >>= 1) s += __shfl_xor(s, o, 32);
  float mean = s * (1.f / 256.f), var = 0.f;
#pragma unroll
  for (int q = 0; q < 8; ++q) { float d = v[q] - mean; var += d * d; }
#pragma unroll
  for (int o = 16; o; o >>= 1) var += __shfl_xor(var, o, 32);
  float rs = rsqrtf(var * (1.f / 256.f) + 1e-5f);
  bf16* y = Y + (size_t)row * 256;
#pragma unroll
  for (int q = 0; q < 8; ++q) {
    int c = lane + 32 * q;
    y[c] = (bf16)((v[q] - mean) * rs * g[c] + b[c]);
  }
}

// qkv (M x 1280 bf16) -> Q[b,h,i,k,d], K1[b,h,i,j,d], K2t[b,h,k,j,d], V1[b,h,i,j,d], V2t[b,h,k,j,d]
__global__ __launch_bounds__(256) void scatter_qkv_kernel(const bf16* __restrict__ qkv,
    bf16* __restrict__ Q, bf16* __restrict__ K1, bf16* __restrict__ K2t,
    bf16* __restrict__ V1, bf16* __restrict__ V2t) {
  size_t idx = (size_t)blockIdx.x * 256 + threadIdx.x;
  int c = (int)(idx % 1280);
  size_t m = idx / 1280;
  int l2 = (int)(m & 127), l1 = (int)((m >> 7) & 127), b = (int)(m >> 14);
  int p = c >> 8, cc = c & 255, hh = cc >> 5, d = cc & 31;
  bf16 val = qkv[idx];
  size_t bh = (size_t)b * 8 + hh;
  size_t o12 = ((bh * 128 + l1) * 128 + l2) * 32 + d;  // (l1,l2)
  size_t o21 = ((bh * 128 + l2) * 128 + l1) * 32 + d;  // (l2,l1)
  switch (p) {
    case 0: Q[o12] = val; break;      // q_ik
    case 1: K1[o12] = val; break;     // k_ij
    case 2: K2t[o21] = val; break;    // k_jk transposed -> [k][j]
    case 3: V1[o12] = val; break;     // v_ij
    default: V2t[o21] = val; break;   // v_jk transposed -> [k][j]
  }
}

// T2[bh,k,i,j] = sum_d Q[bh,i,k,d] * K2t[bh,k,j,d]   (one WMMA k-step: K=HD=32)
__global__ __launch_bounds__(256) void term2_kernel(const bf16* __restrict__ Q,
                                                    const bf16* __restrict__ K2t,
                                                    bf16* __restrict__ T2) {
  const int tid = threadIdx.x, lane = tid & 31, wv = tid >> 5;
  const int k = blockIdx.x & 127;
  const size_t bh = blockIdx.x >> 7;
  __shared__ __align__(16) bf16 sQ[128 * 32];
  __shared__ __align__(16) bf16 sK[128 * 32];
#pragma unroll
  for (int ii = 0; ii < 2; ++ii) {
    int c = tid + 256 * ii;
    int r = c >> 2, cc = c & 3;
    lds_cp16(&Q[((bh * 128 + r) * 128 + k) * 32 + cc * 8], &sQ[r * 32 + cc * 8]);
    lds_cp16(&K2t[(bh * 128 + k) * 4096 + c * 8], &sK[c * 8]);
  }
  async_wait0();
  __syncthreads();
  v16bf fa = frag_a(sQ, 32, wv * 16, 0, lane);
  const int nn = lane & 15, hi = lane >> 4;
  const size_t obase = (bh * 128 + k) * 16384;
#pragma unroll
  for (int c = 0; c < 8; ++c) {
    v8f acc = {};
    v16bf fb = frag_b(sK, 32, c * 16, 0, lane);
    acc = WMMA(fa, fb, acc);
#pragma unroll
    for (int r = 0; r < 8; ++r) {
      int i = wv * 16 + hi * 8 + r, j = c * 16 + nn;
      T2[obase + (size_t)i * 128 + j] = (bf16)acc[r];
    }
  }
}

// Per (bh,i): logits = Q_i K1_i^T + T2[:,i,:]; softmax over j; y1 = P V1_i.
__global__ __launch_bounds__(256) void attn_core_kernel(const bf16* __restrict__ Q,
    const bf16* __restrict__ K1, const bf16* __restrict__ V1, const bf16* __restrict__ T2,
    bf16* __restrict__ P, float* __restrict__ Y) {
  const int tid = threadIdx.x, lane = tid & 31, wv = tid >> 5;
  const int i = blockIdx.x & 127;
  const size_t bh = blockIdx.x >> 7;
  const int b = (int)(bh >> 3), h = (int)(bh & 7);
  __shared__ __align__(16) bf16 sA[128 * 32];
  __shared__ __align__(16) bf16 sB[128 * 32];   // K1 tile, then reused as V1^T (32x128)
  __shared__ __align__(16) float sL[128 * 128];
  __shared__ __align__(16) bf16 sP[128 * 128];
  const size_t rbase = (bh * 128 + i) * 4096;
#pragma unroll
  for (int ii = 0; ii < 2; ++ii) {
    int c = tid + 256 * ii;
    lds_cp16(&Q[rbase + c * 8], &sA[c * 8]);
    lds_cp16(&K1[rbase + c * 8], &sB[c * 8]);
  }
  async_wait0();
  __syncthreads();
  const int nn = lane & 15, hi = lane >> 4;
  const float scale = 0.1767766952966369f;  // 1/sqrt(32)
  {
    v16bf fa = frag_a(sA, 32, wv * 16, 0, lane);
#pragma unroll
    for (int c = 0; c < 8; ++c) {
      v8f acc = {};
      v16bf fb = frag_b(sB, 32, c * 16, 0, lane);
      acc = WMMA(fa, fb, acc);
#pragma unroll
      for (int r = 0; r < 8; ++r) {
        int k = wv * 16 + hi * 8 + r, j = c * 16 + nn;
        float t2 = (float)T2[bh * 2097152 + (size_t)k * 16384 + (size_t)i * 128 + j];
        sL[k * 128 + j] = (acc[r] + t2) * scale;
      }
    }
  }
  __syncthreads();
  // stage V1_i transposed into sB as (d rows x j cols), overlapped with softmax
#pragma unroll
  for (int ii = 0; ii < 16; ++ii) {
    int idx = tid + 256 * ii;
    int j = idx >> 5, d = idx & 31;
    sB[d * 128 + j] = V1[rbase + idx];
  }
  const size_t pbase = (bh * 128 + i) * 16384;
#pragma unroll 1
  for (int rr = 0; rr < 16; ++rr) {
    int k = wv * 16 + rr;
    float e0 = sL[k * 128 + lane],      e1 = sL[k * 128 + 32 + lane];
    float e2 = sL[k * 128 + 64 + lane], e3 = sL[k * 128 + 96 + lane];
    float mx = fmaxf(fmaxf(e0, e1), fmaxf(e2, e3));
#pragma unroll
    for (int o = 16; o; o >>= 1) mx = fmaxf(mx, __shfl_xor(mx, o, 32));
    e0 = __expf(e0 - mx); e1 = __expf(e1 - mx); e2 = __expf(e2 - mx); e3 = __expf(e3 - mx);
    float s = e0 + e1 + e2 + e3;
#pragma unroll
    for (int o = 16; o; o >>= 1) s += __shfl_xor(s, o, 32);
    float inv = 1.f / s;
    bf16 p0 = (bf16)(e0 * inv), p1 = (bf16)(e1 * inv);
    bf16 p2 = (bf16)(e2 * inv), p3 = (bf16)(e3 * inv);
    sP[k * 128 + lane] = p0;      sP[k * 128 + 32 + lane] = p1;
    sP[k * 128 + 64 + lane] = p2; sP[k * 128 + 96 + lane] = p3;
    bf16* pg = P + pbase + (size_t)k * 128;
    pg[lane] = p0; pg[32 + lane] = p1; pg[64 + lane] = p2; pg[96 + lane] = p3;
  }
  __syncthreads();
  v8f acc[2] = {};
#pragma unroll
  for (int kk = 0; kk < 128; kk += 32) {
    v16bf fa = frag_a(sP, 128, wv * 16, kk, lane);
#pragma unroll
    for (int c = 0; c < 2; ++c) {
      v16bf fb = frag_b(sB, 128, c * 16, kk, lane);
      acc[c] = WMMA(fa, fb, acc[c]);
    }
  }
#pragma unroll
  for (int c = 0; c < 2; ++c) {
#pragma unroll
    for (int r = 0; r < 8; ++r) {
      int k = wv * 16 + hi * 8 + r, d = c * 16 + nn;
      Y[(((size_t)b * 16384) + (size_t)i * 128 + k) * 256 + h * 32 + d] = acc[c][r];
    }
  }
}

// Per (bh,k): y2 = P[:,k,:] @ V2t[bh,k]  accumulated into Y (head slice h)
__global__ __launch_bounds__(256) void y2_kernel(const bf16* __restrict__ P,
                                                 const bf16* __restrict__ V2t,
                                                 float* __restrict__ Y) {
  const int tid = threadIdx.x, lane = tid & 31, wv = tid >> 5;
  const int k = blockIdx.x & 127;
  const size_t bh = blockIdx.x >> 7;
  const int b = (int)(bh >> 3), h = (int)(bh & 7);
  __shared__ __align__(16) bf16 sP[128 * 128];
  __shared__ __align__(16) bf16 sV[32 * 128];
#pragma unroll
  for (int ii = 0; ii < 8; ++ii) {
    int c = tid + 256 * ii, r = c >> 4, cc = c & 15;
    lds_cp16(&P[((bh * 128 + r) * 128 + k) * 128 + cc * 8], &sP[r * 128 + cc * 8]);
  }
  const size_t vbase = (bh * 128 + k) * 4096;
#pragma unroll
  for (int ii = 0; ii < 16; ++ii) {   // V2t transposed via VGPRs (scatter, not async-able)
    int idx = tid + 256 * ii;
    int j = idx >> 5, d = idx & 31;
    sV[d * 128 + j] = V2t[vbase + idx];
  }
  async_wait0();
  __syncthreads();
  v8f acc[2] = {};
#pragma unroll
  for (int kk = 0; kk < 128; kk += 32) {
    v16bf fa = frag_a(sP, 128, wv * 16, kk, lane);
#pragma unroll
    for (int c = 0; c < 2; ++c) {
      v16bf fb = frag_b(sV, 128, c * 16, kk, lane);
      acc[c] = WMMA(fa, fb, acc[c]);
    }
  }
  const int nn = lane & 15, hi = lane >> 4;
#pragma unroll
  for (int c = 0; c < 2; ++c) {
#pragma unroll
    for (int r = 0; r < 8; ++r) {
      int i = wv * 16 + hi * 8 + r, d = c * 16 + nn;
      size_t o = (((size_t)b * 16384) + (size_t)i * 128 + k) * 256 + h * 32 + d;
      Y[o] += acc[c][r];
    }
  }
}

extern "C" void kernel_launch(void* const* d_in, const int* in_sizes, int n_in,
                              void* d_out, int out_size, void* d_ws, size_t ws_size,
                              hipStream_t stream) {
  const float* x      = (const float*)d_in[0];
  const float* W_mix  = (const float*)d_in[1];
  const float* W_qkv  = (const float*)d_in[2];
  const float* W_proj = (const float*)d_in[3];
  const float* ln1_g  = (const float*)d_in[4];
  const float* ln1_b  = (const float*)d_in[5];
  const float* ln2_g  = (const float*)d_in[6];
  const float* ln2_b  = (const float*)d_in[7];
  const float* W_ff1  = (const float*)d_in[8];
  const float* W_ff2  = (const float*)d_in[9];
  const float* ffn_s  = (const float*)d_in[10];

  char* ws = (char*)d_ws;
  const size_t MB = 1ull << 20;
  // weights (bf16), persistent
  bf16* wmix_b  = (bf16*)(ws + 0);
  bf16* wqkv_b  = wmix_b + 256 * 256;
  bf16* wproj_b = wqkv_b + 1280 * 256;
  bf16* wff1_b  = wproj_b + 256 * 256;
  bf16* wff2_b  = wff1_b + 1024 * 256;
  // staged buffers (aliased where lifetimes are disjoint; kernels are stream-ordered)
  bf16*  xbt  = (bf16*)(ws + 2 * MB);     // 16.8MB   [xpose..mix]     reused: h2
  float* tmp  = (float*)(ws + 19 * MB);   // 33.5MB   [mix..ln1]       reused: Y (f32)
  bf16*  hbuf = (bf16*)(ws + 53 * MB);    // 16.8MB   [ln1..qkv]       reused: Yb
  bf16*  qkv  = (bf16*)(ws + 70 * MB);    // 83.9MB   [qkv..scatter]   reused: T2 (67MB)
  bf16*  T2   = (bf16*)(ws + 70 * MB);
  bf16*  Qt   = (bf16*)(ws + 154 * MB);   // 5 x 16.8MB               reused: f1 (67MB)
  bf16*  K1t  = Qt + 8388608;
  bf16*  K2t  = K1t + 8388608;
  bf16*  V1t  = K2t + 8388608;
  bf16*  V2t  = V1t + 8388608;
  bf16*  Pbuf = (bf16*)(ws + 238 * MB);   // 67.1MB
  float* x2   = (float*)(ws + 306 * MB);  // 33.5MB  -> total ~340MB
  float* Ybuf = tmp;
  bf16*  Yb   = hbuf;
  bf16*  h2   = xbt;
  bf16*  f1   = Qt;

  // 1) weights + transposed x to bf16
  f32_to_bf16_kernel<<<256, 256, 0, stream>>>(W_mix, wmix_b, 65536);
  f32_to_bf16_kernel<<<1280, 256, 0, stream>>>(W_qkv, wqkv_b, 327680);
  f32_to_bf16_kernel<<<256, 256, 0, stream>>>(W_proj, wproj_b, 65536);
  f32_to_bf16_kernel<<<1024, 256, 0, stream>>>(W_ff1, wff1_b, 262144);
  f32_to_bf16_kernel<<<1024, 256, 0, stream>>>(W_ff2, wff2_b, 262144);
  xpose_x_kernel<<<32768, 256, 0, stream>>>(x, xbt);
  // 2) tmp = x + xT @ W_mix^T
  gemm_bf16_kernel<float, false, true, false><<<dim3(4, 256), 256, 0, stream>>>(
      xbt, wmix_b, tmp, x, nullptr, M_, 256, 256);
  // 3) h = LN1(tmp)
  layernorm_kernel<<<4096, 256, 0, stream>>>(tmp, hbuf, ln1_g, ln1_b);
  // 4) qkv = h @ W_qkv^T ; scatter to head-major layouts
  gemm_bf16_kernel<bf16, false, false, false><<<dim3(20, 256), 256, 0, stream>>>(
      hbuf, wqkv_b, qkv, nullptr, nullptr, M_, 1280, 256);
  scatter_qkv_kernel<<<163840, 256, 0, stream>>>(qkv, Qt, K1t, K2t, V1t, V2t);
  // 5) attention
  term2_kernel<<<2048, 256, 0, stream>>>(Qt, K2t, T2);
  attn_core_kernel<<<2048, 256, 0, stream>>>(Qt, K1t, V1t, T2, Pbuf, Ybuf);
  y2_kernel<<<2048, 256, 0, stream>>>(Pbuf, V2t, Ybuf);
  // 6) x2 = x + Y @ W_proj^T
  f32_to_bf16_kernel<<<32768, 256, 0, stream>>>(Ybuf, Yb, 8388608);
  gemm_bf16_kernel<float, false, true, false><<<dim3(4, 256), 256, 0, stream>>>(
      Yb, wproj_b, x2, x, nullptr, M_, 256, 256);
  // 7) FFN: out = x2 + ffn_scale * (relu(LN2(x2) @ W_ff1^T) @ W_ff2^T)
  layernorm_kernel<<<4096, 256, 0, stream>>>(x2, h2, ln2_g, ln2_b);
  gemm_bf16_kernel<bf16, true, false, false><<<dim3(16, 256), 256, 0, stream>>>(
      h2, wff1_b, f1, nullptr, nullptr, M_, 1024, 256);
  gemm_bf16_kernel<float, false, true, true><<<dim3(4, 256), 256, 0, stream>>>(
      f1, wff2_b, (float*)d_out, x2, ffn_s, M_, 256, 1024);
}
